// BaseDecoder_68702296867024
// MI455X (gfx1250) — compile-verified, matrix operand order checked by reading, not run on hardware
//
#include <hip/hip_runtime.h>
#include <math.h>

typedef __attribute__((ext_vector_type(16))) _Float16 v16h;
typedef __attribute__((ext_vector_type(8)))  _Float16 v8h;
typedef __attribute__((ext_vector_type(4)))  _Float16 v4h;
typedef __attribute__((ext_vector_type(8)))  float    v8f;

#define WMMA_F16(a, b, c) \
    __builtin_amdgcn_wmma_f32_16x16x32_f16(false, (a), false, (b), (short)0, (c), false, false)

// LDS byte offset of a generic pointer that points into LDS (AS3 value == offset)
__device__ __forceinline__ unsigned lds_off(const void* p) {
    return (unsigned)(unsigned long long)(const __attribute__((address_space(3))) void*)p;
}

// Per-lane async DMA: 16 bytes global -> LDS, tracked by ASYNCcnt.
__device__ __forceinline__ void async_copy16(const void* gsrc, const void* ldst) {
    asm volatile("global_load_async_to_lds_b128 %0, %1, off"
                 :: "v"(lds_off(ldst)), "v"(gsrc) : "memory");
}
__device__ __forceinline__ void wait_async0() {
    asm volatile("s_wait_asynccnt 0x0" ::: "memory");
}

// Load a 16-half WMMA fragment for one lane.
// Layout (CDNA5 ISA 7.12.2, 16-bit A/B): lane%16 selects row (A) / col (B);
// element e maps to K = (e<8 ? e : e+8) + 8*(lane/16). Two contiguous 16B chunks.
__device__ __forceinline__ v16h load_frag(const _Float16* __restrict__ rowbase, int hl) {
    v8h a0 = *(const v8h*)(rowbase + 8 * hl);
    v8h a1 = *(const v8h*)(rowbase + 16 + 8 * hl);
    v16h r;
#pragma unroll
    for (int e = 0; e < 8; ++e) { r[e] = a0[e]; r[e + 8] = a1[e]; }
    return r;
}

// ---------------------------------------------------------------------------
// GEMM: Y[M,N] = act(X[M,K] @ W[N,K]^T + bias[N]); X,W f16.
// O16: Y stored f16 else f32. ASYNC: double-buffered global->LDS async DMA
// (requires M%128==0, N%64==0); else guarded synchronous staging.
// Block tile 128x64, 8 waves, each wave 32x32 (2x2 WMMA 16x16x32 tiles).
// ---------------------------------------------------------------------------
#define BM 128
#define BN 64
#define BK 32
#define ASTR 48   // halfs per LDS row (bank-conflict padding, 16B aligned)

template <bool O16, bool GELU_ACT, bool ASYNC>
__global__ __launch_bounds__(256) void gemm_kernel(
    const _Float16* __restrict__ A, const _Float16* __restrict__ W,
    const float* __restrict__ bias, void* __restrict__ Out,
    int M, int N, int K)
{
    __shared__ _Float16 As[2][BM * ASTR];
    __shared__ _Float16 Bs[2][BN * ASTR];

    const int tid  = threadIdx.x;
    const int wave = tid >> 5;
    const int lane = tid & 31;
    const int hl   = lane >> 4;
    const int l16  = lane & 15;
    const int wm   = wave >> 1;   // 0..3  (32-row band)
    const int wn   = wave & 1;    // 0..1  (32-col band)
    const long Mb  = (long)blockIdx.x * BM;
    const long Nb  = (long)blockIdx.y * BN;
    const int  nk  = K / BK;

    // staging coordinates: 16B chunks of a row; row stride K halfs in global
    const int crow = tid >> 2;            // 0..63
    const int ccol = (tid & 3) << 3;      // halfs: 0,8,16,24

    v8f acc[2][2];
#pragma unroll
    for (int i = 0; i < 2; ++i)
#pragma unroll
        for (int j = 0; j < 2; ++j)
#pragma unroll
            for (int r = 0; r < 8; ++r) acc[i][j][r] = 0.0f;

    if (ASYNC) {
        // ---- issue tile kb into buffer bi (3 async b128 ops per thread) ----
        auto issue = [&](int kb, int bi) {
            const long k0 = (long)kb * BK + ccol;
            async_copy16(A + (Mb + crow) * (long)K + k0,
                         &As[bi][crow * ASTR + ccol]);
            async_copy16(A + (Mb + 64 + crow) * (long)K + k0,
                         &As[bi][(64 + crow) * ASTR + ccol]);
            async_copy16(W + (Nb + crow) * (long)K + k0,
                         &Bs[bi][crow * ASTR + ccol]);
        };
        issue(0, 0);
        for (int kb = 0; kb < nk; ++kb) {
            const int bi = kb & 1;
            wait_async0();        // tile kb resident (async loads complete in order)
            __syncthreads();      // all waves see tile kb; prev compute done
            if (kb + 1 < nk) issue(kb + 1, bi ^ 1);   // overlap DMA with WMMA

            v16h a0 = load_frag(&As[bi][(wm * 32 + l16) * ASTR], hl);
            v16h a1 = load_frag(&As[bi][(wm * 32 + 16 + l16) * ASTR], hl);
            v16h b0 = load_frag(&Bs[bi][(wn * 32 + l16) * ASTR], hl);
            v16h b1 = load_frag(&Bs[bi][(wn * 32 + 16 + l16) * ASTR], hl);
            acc[0][0] = WMMA_F16(a0, b0, acc[0][0]);
            acc[0][1] = WMMA_F16(a0, b1, acc[0][1]);
            acc[1][0] = WMMA_F16(a1, b0, acc[1][0]);
            acc[1][1] = WMMA_F16(a1, b1, acc[1][1]);
        }
    } else {
        for (int kb = 0; kb < nk; ++kb) {
            __syncthreads();
            const long k0 = (long)kb * BK + ccol;
            // A rows (guarded)
#pragma unroll
            for (int rr = 0; rr < BM; rr += 64) {
                long gm = Mb + rr + crow;
                v8h st;
                if (gm < M) st = *(const v8h*)(A + gm * (long)K + k0);
                else {
#pragma unroll
                    for (int e = 0; e < 8; ++e) st[e] = (_Float16)0.0f;
                }
                *(v8h*)&As[0][(rr + crow) * ASTR + ccol] = st;
            }
            // W rows (guarded)
            {
                long gn = Nb + crow;
                v8h st;
                if (gn < N) st = *(const v8h*)(W + gn * (long)K + k0);
                else {
#pragma unroll
                    for (int e = 0; e < 8; ++e) st[e] = (_Float16)0.0f;
                }
                *(v8h*)&Bs[0][crow * ASTR + ccol] = st;
            }
            __syncthreads();

            v16h a0 = load_frag(&As[0][(wm * 32 + l16) * ASTR], hl);
            v16h a1 = load_frag(&As[0][(wm * 32 + 16 + l16) * ASTR], hl);
            v16h b0 = load_frag(&Bs[0][(wn * 32 + l16) * ASTR], hl);
            v16h b1 = load_frag(&Bs[0][(wn * 32 + 16 + l16) * ASTR], hl);
            acc[0][0] = WMMA_F16(a0, b0, acc[0][0]);
            acc[0][1] = WMMA_F16(a0, b1, acc[0][1]);
            acc[1][0] = WMMA_F16(a1, b0, acc[1][0]);
            acc[1][1] = WMMA_F16(a1, b1, acc[1][1]);
        }
    }

    // ---- store (D layout: lane%16 = col, VGPR r = row r + 8*(lane/16)) ----
#pragma unroll
    for (int tm = 0; tm < 2; ++tm) {
#pragma unroll
        for (int tn = 0; tn < 2; ++tn) {
            long gn  = Nb + wn * 32 + tn * 16 + l16;
            float bv = (gn < N) ? bias[gn] : 0.0f;
#pragma unroll
            for (int r = 0; r < 8; ++r) {
                long gm = Mb + wm * 32 + tm * 16 + r + 8 * hl;
                if (gm < M && gn < N) {
                    float v = acc[tm][tn][r] + bv;
                    if (GELU_ACT) v = 0.5f * v * (1.0f + erff(v * 0.70710678118f));
                    if (O16) ((_Float16*)Out)[gm * (long)N + gn] = (_Float16)v;
                    else     ((float*)Out)[gm * (long)N + gn] = v;
                }
            }
        }
    }
}

// ---------------------------------------------------------------------------
// Attention: one block per (batch, head, 16-query tile). Lk == 512.
// mode 0: self (causal + distance bias + target pad mask)
// mode 1: cross (memory key pad mask)
// ---------------------------------------------------------------------------
__global__ __launch_bounds__(256) void attn_kernel(
    const _Float16* __restrict__ Qp, int qs,
    const _Float16* __restrict__ Kp, int ks,
    const _Float16* __restrict__ Vp, int vs,
    _Float16* __restrict__ Op,
    int Lk, int mode,
    const int* __restrict__ dsq, const float* __restrict__ demb,
    const int* __restrict__ seq, const unsigned char* __restrict__ mmask)
{
    extern __shared__ char smem[];
    float*    S   = (float*)smem;                    // [16][512] f32 scores
    _Float16* P   = (_Float16*)(smem + 32768);       // [16][520] f16 probs
    _Float16* Vt  = (_Float16*)(smem + 49408);       // [64][520] f16 V^T
    float*    red = (float*)(smem + 115968);         // [4][32][8] split-K partials

    const int tid  = threadIdx.x;
    const int wave = tid >> 5;
    const int lane = tid & 31;
    const int hl   = lane >> 4;
    const int l16  = lane & 15;
    const int qt   = blockIdx.x & 31;
    const int h    = (blockIdx.x >> 5) & 15;
    const int b    = blockIdx.x >> 9;

    const long qrow = (long)b * 512 + qt * 16 + l16;
    const _Float16* qbase = Qp + qrow * (long)qs + h * 64;
    v16h q0 = load_frag(qbase, hl);
    v16h q1 = load_frag(qbase + 32, hl);

    // ---- S = (Q K^T) * scale + bias ----
    for (int j = 0; j < 4; ++j) {
        int kt = wave * 4 + j;
        const long krow = (long)b * Lk + kt * 16 + l16;
        const _Float16* kbase = Kp + krow * (long)ks + h * 64;
        v16h k0 = load_frag(kbase, hl);
        v16h k1 = load_frag(kbase + 32, hl);
        v8f acc;
#pragma unroll
        for (int r = 0; r < 8; ++r) acc[r] = 0.0f;
        acc = WMMA_F16(q0, k0, acc);
        acc = WMMA_F16(q1, k1, acc);

        int key = kt * 16 + l16;
#pragma unroll
        for (int r = 0; r < 8; ++r) {
            int m  = r + 8 * hl;
            int qi = qt * 16 + m;
            float s = acc[r] * 0.125f;   // 1/sqrt(64)
            if (mode == 0) {
                if (key <= qi) s += demb[dsq[((long)b * 512 + qi) * 512 + key] * 16 + h];
                else           s += -1e9f;
                if (seq[b * 512 + key] == 0) s += -1e9f;
            } else {
                if (mmask[b * Lk + key]) s += -1e9f;
            }
            S[m * 512 + key] = s;
        }
    }
    __syncthreads();

    // ---- softmax, 2 rows per wave ----
#pragma unroll
    for (int rr = 0; rr < 2; ++rr) {
        int m = wave * 2 + rr;
        float mx = -3.4e38f;
        for (int i = lane; i < 512; i += 32) mx = fmaxf(mx, S[m * 512 + i]);
#pragma unroll
        for (int off = 16; off >= 1; off >>= 1) mx = fmaxf(mx, __shfl_xor(mx, off, 32));
        float sum = 0.0f;
        for (int i = lane; i < 512; i += 32) {
            float e = __expf(S[m * 512 + i] - mx);
            S[m * 512 + i] = e;
            sum += e;
        }
#pragma unroll
        for (int off = 16; off >= 1; off >>= 1) sum += __shfl_xor(sum, off, 32);
        float inv = 1.0f / sum;
        for (int i = lane; i < 512; i += 32)
            P[m * 520 + i] = (_Float16)(S[m * 512 + i] * inv);
    }

    // ---- stage V^T into LDS: Vt[d][key] ----
    {
        int d4 = (tid & 15) << 2;
        for (int key = tid >> 4; key < Lk; key += 16) {
            v4h vv = *(const v4h*)(Vp + ((long)b * Lk + key) * (long)vs + h * 64 + d4);
#pragma unroll
            for (int i = 0; i < 4; ++i) Vt[(d4 + i) * 520 + key] = vv[i];
        }
    }
    __syncthreads();

    // ---- O = P @ V : 4 d-tiles x split-K over 2 wave groups ----
    const int dt = wave & 3;
    const int kh = wave >> 2;
    v8f acc;
#pragma unroll
    for (int r = 0; r < 8; ++r) acc[r] = 0.0f;
    for (int kb = kh * 8; kb < kh * 8 + 8; ++kb) {
        v16h a  = load_frag(&P[l16 * 520 + kb * 32], hl);
        v16h bb = load_frag(&Vt[(dt * 16 + l16) * 520 + kb * 32], hl);
        acc = WMMA_F16(a, bb, acc);
    }
    if (kh == 1) {
#pragma unroll
        for (int r = 0; r < 8; ++r) red[dt * 256 + lane * 8 + r] = acc[r];
    }
    __syncthreads();
    if (kh == 0) {
#pragma unroll
        for (int r = 0; r < 8; ++r) {
            float v = acc[r] + red[dt * 256 + lane * 8 + r];
            int m = r + 8 * hl;
            Op[((long)b * 512 + qt * 16 + m) * 1024 + h * 64 + dt * 16 + l16] = (_Float16)v;
        }
    }
}

// ---------------------------------------------------------------------------
// Fused residual + LayerNorm over E=1024. One wave per row.
// out = LN(acc + res); also writes f16 mirror when out16 != null.
// ---------------------------------------------------------------------------
__global__ __launch_bounds__(256) void ln_kernel(
    const float* __restrict__ acc, const float* __restrict__ res,
    const float* __restrict__ g, const float* __restrict__ bta,
    float* __restrict__ out, _Float16* __restrict__ out16, int rows)
{
    const int wave = threadIdx.x >> 5;
    const int lane = threadIdx.x & 31;
    const long row = (long)blockIdx.x * 8 + wave;
    if (row >= rows) return;

    const float* ap = acc + row * 1024;
    const float* rp = res ? res + row * 1024 : nullptr;
    float v[32];
    float sum = 0.0f;
#pragma unroll
    for (int j = 0; j < 32; ++j) {
        int e = j * 32 + lane;
        float x = ap[e] + (rp ? rp[e] : 0.0f);
        v[j] = x;
        sum += x;
    }
#pragma unroll
    for (int off = 16; off >= 1; off >>= 1) sum += __shfl_xor(sum, off, 32);
    float mean = sum * (1.0f / 1024.0f);
    float vs = 0.0f;
#pragma unroll
    for (int j = 0; j < 32; ++j) { float d = v[j] - mean; vs += d * d; }
#pragma unroll
    for (int off = 16; off >= 1; off >>= 1) vs += __shfl_xor(vs, off, 32);
    float rstd = rsqrtf(vs * (1.0f / 1024.0f) + 1e-5f);
#pragma unroll
    for (int j = 0; j < 32; ++j) {
        int e = j * 32 + lane;
        float o = (v[j] - mean) * rstd * g[e] + bta[e];
        out[row * 1024 + e] = o;
        if (out16) out16[row * 1024 + e] = (_Float16)o;
    }
}

// x[b,t,:] = tok_emb[seq[b,t]] * sqrt(E); writes f32 and f16 mirrors
__global__ __launch_bounds__(256) void embed_kernel(
    const int* __restrict__ seq, const float* __restrict__ emb,
    float* __restrict__ x, _Float16* __restrict__ x16)
{
    long i = (long)blockIdx.x * 256 + threadIdx.x;   // over 8192*1024
    long row = i >> 10;
    long col = i & 1023;
    float v = emb[(long)seq[row] * 1024 + col] * 32.0f;
    x[i] = v;
    x16[i] = (_Float16)v;
}

// f32 -> f16 bulk convert (n % 4 == 0)
__global__ __launch_bounds__(256) void cvt_kernel(
    const float* __restrict__ s, _Float16* __restrict__ d, long n)
{
    long i = ((long)blockIdx.x * 256 + threadIdx.x) * 4;
    if (i < n) {
        float4 v = *(const float4*)(s + i);
        v4h o;
        o[0] = (_Float16)v.x; o[1] = (_Float16)v.y;
        o[2] = (_Float16)v.z; o[3] = (_Float16)v.w;
        *(v4h*)(d + i) = o;
    }
}

// ---------------------------------------------------------------------------
extern "C" void kernel_launch(void* const* d_in, const int* in_sizes, int n_in,
                              void* d_out, int out_size, void* d_ws, size_t ws_size,
                              hipStream_t stream)
{
    const int*   seq      = (const int*)d_in[0];
    const int*   dsq      = (const int*)d_in[1];
    const float* memory   = (const float*)d_in[2];
    const unsigned char* mmask = (const unsigned char*)d_in[3];
    const float* tok_emb  = (const float*)d_in[4];
    const float* demb     = (const float*)d_in[5];
    const float* self_in_w  = (const float*)d_in[6];
    const float* self_in_b  = (const float*)d_in[7];
    const float* self_out_w = (const float*)d_in[8];
    const float* self_out_b = (const float*)d_in[9];
    const float* cross_in_w  = (const float*)d_in[10];
    const float* cross_in_b  = (const float*)d_in[11];
    const float* cross_out_w = (const float*)d_in[12];
    const float* cross_out_b = (const float*)d_in[13];
    const float* ffn_w1 = (const float*)d_in[14];
    const float* ffn_b1 = (const float*)d_in[15];
    const float* ffn_w2 = (const float*)d_in[16];
    const float* ffn_b2 = (const float*)d_in[17];
    const float* ln1_g = (const float*)d_in[18];
    const float* ln1_b = (const float*)d_in[19];
    const float* ln2_g = (const float*)d_in[20];
    const float* ln2_b = (const float*)d_in[21];
    const float* ln3_g = (const float*)d_in[22];
    const float* ln3_b = (const float*)d_in[23];
    const float* normf_g = (const float*)d_in[24];
    const float* normf_b = (const float*)d_in[25];
    const float* gen_w = (const float*)d_in[26];
    const float* gen_b = (const float*)d_in[27];
    float* out = (float*)d_out;

    // ---- workspace layout ----
    char* w = (char*)d_ws;
    auto alloc_f32 = [&](size_t n) { float* p = (float*)w;    w += n * 4; return p; };
    auto alloc_f16 = [&](size_t n) { _Float16* p = (_Float16*)w; w += n * 2; return p; };

    float*    x    = alloc_f32((size_t)8192 * 1024);   // activations f32 (residual chain)
    _Float16* x16  = alloc_f16((size_t)8192 * 1024);   // f16 mirror
    _Float16* qkv  = alloc_f16((size_t)8192 * 3072);   // QKV / cross q+kv
    _Float16* ao   = alloc_f16((size_t)8192 * 1024);   // attention out
    float*    yb   = alloc_f32((size_t)8192 * 1024);   // pre-LN f32
    _Float16* yb16 = alloc_f16((size_t)8192 * 1024);   // final-LN f16
    _Float16* hb   = alloc_f16((size_t)8192 * 4096);   // FFN hidden
    _Float16* mem16 = alloc_f16((size_t)8192 * 1024);  // encoder memory f16
    // f16 weights
    _Float16* wsi = alloc_f16((size_t)6 * 3145728);
    _Float16* wso = alloc_f16((size_t)6 * 1048576);
    _Float16* wci = alloc_f16((size_t)6 * 3145728);
    _Float16* wco = alloc_f16((size_t)6 * 1048576);
    _Float16* wf1 = alloc_f16((size_t)6 * 4194304);
    _Float16* wf2 = alloc_f16((size_t)6 * 4194304);
    _Float16* wgn = alloc_f16((size_t)70 * 1024);
    _Float16* q16  = qkv;
    _Float16* kv16 = qkv + (size_t)8192 * 1024;

    auto cvt = [&](const float* s, _Float16* d, size_t n) {
        cvt_kernel<<<(unsigned)((n / 4 + 255) / 256), 256, 0, stream>>>(s, d, (long)n);
    };
    cvt(memory,      mem16, (size_t)8192 * 1024);
    cvt(self_in_w,   wsi,   (size_t)6 * 3145728);
    cvt(self_out_w,  wso,   (size_t)6 * 1048576);
    cvt(cross_in_w,  wci,   (size_t)6 * 3145728);
    cvt(cross_out_w, wco,   (size_t)6 * 1048576);
    cvt(ffn_w1,      wf1,   (size_t)6 * 4194304);
    cvt(ffn_w2,      wf2,   (size_t)6 * 4194304);
    cvt(gen_w,       wgn,   (size_t)70 * 1024);

    const int SMEM = 120064;  // S(32768) + P(16640) + Vt(66560) + red(4096)

    embed_kernel<<<32768, 256, 0, stream>>>(seq, tok_emb, x, x16);

    for (int i = 0; i < 6; ++i) {
        // ---- self attention ----
        gemm_kernel<true, false, true><<<dim3(64, 48), 256, 0, stream>>>(
            x16, wsi + (size_t)i * 3145728, self_in_b + (size_t)i * 3072, qkv,
            8192, 3072, 1024);
        attn_kernel<<<8192, 256, SMEM, stream>>>(
            qkv, 3072, qkv + 1024, 3072, qkv + 2048, 3072, ao,
            512, 0, dsq, demb, seq, nullptr);
        gemm_kernel<false, false, true><<<dim3(64, 16), 256, 0, stream>>>(
            ao, wso + (size_t)i * 1048576, self_out_b + (size_t)i * 1024, yb,
            8192, 1024, 1024);
        ln_kernel<<<1024, 256, 0, stream>>>(yb, x, ln1_g + (size_t)i * 1024,
                                            ln1_b + (size_t)i * 1024, x, x16, 8192);

        // ---- cross attention ----
        gemm_kernel<true, false, true><<<dim3(64, 16), 256, 0, stream>>>(
            x16, wci + (size_t)i * 3145728, cross_in_b + (size_t)i * 3072, q16,
            8192, 1024, 1024);
        gemm_kernel<true, false, true><<<dim3(64, 32), 256, 0, stream>>>(
            mem16, wci + (size_t)i * 3145728 + 1048576,
            cross_in_b + (size_t)i * 3072 + 1024, kv16,
            8192, 2048, 1024);
        attn_kernel<<<8192, 256, SMEM, stream>>>(
            q16, 1024, kv16, 2048, kv16 + 1024, 2048, ao,
            512, 1, nullptr, nullptr, nullptr, mmask);
        gemm_kernel<false, false, true><<<dim3(64, 16), 256, 0, stream>>>(
            ao, wco + (size_t)i * 1048576, cross_out_b + (size_t)i * 1024, yb,
            8192, 1024, 1024);
        ln_kernel<<<1024, 256, 0, stream>>>(yb, x, ln2_g + (size_t)i * 1024,
                                            ln2_b + (size_t)i * 1024, x, x16, 8192);

        // ---- FFN ----
        gemm_kernel<true, true, true><<<dim3(64, 64), 256, 0, stream>>>(
            x16, wf1 + (size_t)i * 4194304, ffn_b1 + (size_t)i * 4096, hb,
            8192, 4096, 1024);
        gemm_kernel<false, false, true><<<dim3(64, 16), 256, 0, stream>>>(
            hb, wf2 + (size_t)i * 4194304, ffn_b2 + (size_t)i * 1024, yb,
            8192, 1024, 4096);
        ln_kernel<<<1024, 256, 0, stream>>>(yb, x, ln3_g + (size_t)i * 1024,
                                            ln3_b + (size_t)i * 1024, x, x16, 8192);
    }

    // final LN + generator (N=70 -> guarded synchronous staging path)
    ln_kernel<<<1024, 256, 0, stream>>>(x, nullptr, normf_g, normf_b, yb, yb16, 8192);
    gemm_kernel<false, false, false><<<dim3(64, 2), 256, 0, stream>>>(
        yb16, wgn, gen_b, out, 8192, 70, 1024);
}
